// ImplicitRNNCell_28681791603310
// MI455X (gfx1250) — matrix operand order repaired and checked by reading, not compiled, not run on hardware
//
#include <hip/hip_runtime.h>

// Implicit RNN (Picard fixed-point) for MI455X / gfx1250.
// Persistent kernel: 16 WGs x 8 waves = 128 waves; one 16x16 WMMA tile per wave
// over the 4(M=64) x 32(N=512) tile grid. bf16 v_wmma_f32_16x16x32_bf16 path,
// f32 accumulate. A-weight fragments pinned in VGPRs for the whole kernel.
// Fragment loads are issued as one un-splittable burst; the accumulator is
// threaded through an asm wait so the WMMA chain cannot be hoisted above it:
// each fixed-point GEMM pays ONE L2 round trip + 16 back-to-back WMMAs.

#define T_SEQ 1024
#define BATCH 64
#define IDIM  64
#define HDIM  256
#define NDIM  512
#define PDIM  (IDIM + HDIM)   // 320
#define NFP   20
#define NWG   16
#define TPB   256
#define KN    (NDIM / 32)     // 16 k-steps for K=512
#define KP    (PDIM / 32)     // 10 k-steps for K=320

typedef __bf16 bf16_t;
typedef __attribute__((ext_vector_type(16))) __bf16 v16bf;
typedef __attribute__((ext_vector_type(8)))  __bf16 v8bf;
typedef __attribute__((ext_vector_type(8)))  float  v8f;

// Scheduling fence threaded through the accumulator: loads above cannot sink
// (memory clobber), and the WMMA chain below cannot hoist (data dep on acc).
#define MMA_FENCE(accv) asm volatile("s_wait_loadcnt 0x0" : "+v"(accv) :: "memory")

__device__ inline v8f zero8() {
  v8f r;
#pragma unroll
  for (int i = 0; i < 8; ++i) r[i] = 0.f;
  return r;
}

// A-fragment: 16x32 bf16 tile from row-major [rows x ldk].
// Contiguous k order (same as load_bfrag): lanes 0-15 row M=lane, K {0..15}; lanes 16-31 K {16..31}.
__device__ inline v16bf load_afrag(const bf16_t* __restrict__ base, int row0, int ldk,
                                   int k0, int lane) {
  const int r  = row0 + (lane & 15);
  const int hi = (lane >> 4) & 1;
  const bf16_t* p = base + (size_t)r * ldk + k0 + hi * 16;
  v8bf lo = *(const v8bf*)p;          // K = k0 + hi*16 + 0..7
  v8bf hv = *(const v8bf*)(p + 8);    // K = k0 + hi*16 + 8..15
  v16bf a;
#pragma unroll
  for (int i = 0; i < 8; ++i) { a[i] = lo[i]; a[i + 8] = hv[i]; }
  return a;
}

// B-fragment: 32x16 bf16 where B = W^T tile. Column n of B == row (col0+n) of
// row-major W[N x ldk] -> one contiguous 32B load per lane.
__device__ inline v16bf load_bfrag(const bf16_t* __restrict__ base, int col0, int ldk,
                                   int k0, int lane) {
  const int n  = col0 + (lane & 15);
  const int hi = (lane >> 4) & 1;
  return *(const v16bf*)(base + (size_t)n * ldk + k0 + hi * 16);
}

// f32 C/D accumulator layout: VGPR i -> row row0 + i + (lane>=16 ? 8:0), col = col0 + lane%16.
__device__ inline void store_tile_relu_bf16(bf16_t* __restrict__ dst, int row0, int col0,
                                            int ld, v8f acc, int lane) {
  const int c  = col0 + (lane & 15);
  const int hi = (lane >> 4) & 1;
#pragma unroll
  for (int i = 0; i < 8; ++i) {
    float v = fmaxf(acc[i], 0.f);
    dst[(size_t)(row0 + i + hi * 8) * ld + c] = (bf16_t)v;
  }
}

// Device-scope sense-reversal barrier across the 16 persistent workgroups.
__device__ inline void gbar(unsigned* cnt, unsigned* gen) {
  __threadfence();     // flush this thread's stores to device scope (L2)
  __syncthreads();     // whole WG's fences complete before we signal
  if (threadIdx.x == 0) {
    unsigned g = __hip_atomic_load(gen, __ATOMIC_ACQUIRE, __HIP_MEMORY_SCOPE_AGENT);
    unsigned arrived =
        __hip_atomic_fetch_add(cnt, 1u, __ATOMIC_ACQ_REL, __HIP_MEMORY_SCOPE_AGENT) + 1u;
    if (arrived == (unsigned)NWG) {
      __hip_atomic_store(cnt, 0u, __ATOMIC_RELAXED, __HIP_MEMORY_SCOPE_AGENT);
      __hip_atomic_fetch_add(gen, 1u, __ATOMIC_RELEASE, __HIP_MEMORY_SCOPE_AGENT);
    } else {
      while (__hip_atomic_load(gen, __ATOMIC_ACQUIRE, __HIP_MEMORY_SCOPE_AGENT) == g)
        __builtin_amdgcn_s_sleep(2);
    }
  }
  __syncthreads();
  __threadfence();     // acquire side: invalidate stale near caches
}

__global__ void init_barrier_kernel(unsigned* cnt, unsigned* gen) {
  if (threadIdx.x == 0) { *cnt = 0u; *gen = 0u; }
}

__global__ void cvt_f32_bf16_kernel(const float* __restrict__ src, bf16_t* __restrict__ dst,
                                    int n) {
  int i = blockIdx.x * blockDim.x + threadIdx.x;
  if (i < n) dst[i] = (bf16_t)src[i];
}

__global__ __launch_bounds__(TPB)
void implicit_rnn_kernel(const float* __restrict__ x, float* __restrict__ out,
                         const bf16_t* __restrict__ Aw, const bf16_t* __restrict__ Bw,
                         const bf16_t* __restrict__ Cw, const bf16_t* __restrict__ Dw,
                         bf16_t* __restrict__ u, bf16_t* __restrict__ zbuf,
                         bf16_t* __restrict__ h, unsigned* cnt, unsigned* gen) {
  const int lane    = threadIdx.x & 31;
  const int wave    = blockIdx.x * (TPB / 32) + (threadIdx.x >> 5);  // 0..127
  const int gtid    = blockIdx.x * TPB + threadIdx.x;
  const int gstride = NWG * TPB;

  const int mz = wave & 3;   // M tile (batch rows, 4 blocks of 16)
  const int nz = wave >> 2;  // N tile for z phase (32 blocks of 16)
  const int no = wave >> 2;  // N tile for output phase (valid when wave < 64)

  bf16_t* z0 = zbuf;
  bf16_t* z1 = zbuf + (size_t)BATCH * NDIM;

  // h0 = 0
  for (int e = gtid; e < BATCH * HDIM; e += gstride) h[e] = (bf16_t)0.f;

  // Pin the A-weight B-fragments for this wave's N-block in VGPRs for the
  // entire kernel: reused 19x per timestep x 1024 steps. 16 frags = 128 VGPRs.
  v16bf Af[KN];
#pragma unroll
  for (int kk = 0; kk < KN; ++kk) Af[kk] = load_bfrag(Aw, nz * 16, NDIM, kk * 32, lane);

  gbar(cnt, gen);

#pragma unroll 1
  for (int t = 0; t < T_SEQ; ++t) {
    // u = [x_t | h]  (BATCH x PDIM, bf16)
#pragma unroll 1
    for (int e = gtid; e < BATCH * PDIM; e += gstride) {
      int b = e / PDIM, j = e - b * PDIM;
      float v = (j < IDIM) ? x[(size_t)b * T_SEQ * IDIM + (size_t)t * IDIM + j]
                           : (float)h[b * HDIM + (j - IDIM)];
      u[e] = (bf16_t)v;
    }
    gbar(cnt, gen);

    // ---- Bu tile: one load burst, fence, then chained WMMAs ----
    v8f bu;
    {
      v16bf uf[KP], bf[KP];
#pragma unroll
      for (int kk = 0; kk < KP; ++kk) {
        uf[kk] = load_afrag(u, mz * 16, PDIM, kk * 32, lane);
        bf[kk] = load_bfrag(Bw, nz * 16, PDIM, kk * 32, lane);
      }
      bu = zero8();
      MMA_FENCE(bu);
#pragma unroll
      for (int kk = 0; kk < KP; ++kk)
        bu = __builtin_amdgcn_wmma_f32_16x16x32_bf16(false, uf[kk], false, bf[kk],
                                                     (short)0, bu, false, false);
    }
    // Iteration 1: z = relu(0*A + Bu) = relu(Bu)
    store_tile_relu_bf16(z0, mz * 16, nz * 16, NDIM, bu, lane);
    gbar(cnt, gen);

    // ---- Iterations 2..20, double-buffered z (bf16, L2-resident) ----
#pragma unroll 1
    for (int it = 1; it < NFP; ++it) {
      const bf16_t* zr = (it & 1) ? z0 : z1;
      bf16_t*       zw = (it & 1) ? z1 : z0;
      // Burst: all 32 b128 loads of this row block issued back-to-back, one
      // wait, then 16 back-to-back WMMAs against the pinned Af fragments.
      v16bf zf[KN];
#pragma unroll
      for (int kk = 0; kk < KN; ++kk)
        zf[kk] = load_afrag(zr, mz * 16, NDIM, kk * 32, lane);
      v8f acc = bu;
      MMA_FENCE(acc);
#pragma unroll
      for (int kk = 0; kk < KN; ++kk)
        acc = __builtin_amdgcn_wmma_f32_16x16x32_bf16(false, zf[kk], false, Af[kk],
                                                      (short)0, acc, false, false);
      store_tile_relu_bf16(zw, mz * 16, nz * 16, NDIM, acc, lane);
      gbar(cnt, gen);
    }

    // ---- Output: h_new = z@C^T + u@D^T  (final z in z1: it=19 is odd) ----
    if (wave < 64) {
      v8f acc = zero8();
      {
        v16bf zf[KN], cf[KN];
#pragma unroll
        for (int kk = 0; kk < KN; ++kk) {
          zf[kk] = load_afrag(z1, mz * 16, NDIM, kk * 32, lane);
          cf[kk] = load_bfrag(Cw, no * 16, NDIM, kk * 32, lane);
        }
        MMA_FENCE(acc);
#pragma unroll
        for (int kk = 0; kk < KN; ++kk)
          acc = __builtin_amdgcn_wmma_f32_16x16x32_bf16(false, zf[kk], false, cf[kk],
                                                        (short)0, acc, false, false);
      }
      {
        v16bf uf[KP], df[KP];
#pragma unroll
        for (int kk = 0; kk < KP; ++kk) {
          uf[kk] = load_afrag(u, mz * 16, PDIM, kk * 32, lane);
          df[kk] = load_bfrag(Dw, no * 16, PDIM, kk * 32, lane);
        }
        MMA_FENCE(acc);
#pragma unroll
        for (int kk = 0; kk < KP; ++kk)
          acc = __builtin_amdgcn_wmma_f32_16x16x32_bf16(false, uf[kk], false, df[kk],
                                                        (short)0, acc, false, false);
      }
      const int c  = no * 16 + (lane & 15);
      const int hi = (lane >> 4) & 1;
#pragma unroll
      for (int i = 0; i < 8; ++i) {
        const int   bb = mz * 16 + i + hi * 8;
        const float v  = acc[i];
        h[bb * HDIM + c] = (bf16_t)v;                                   // recurrent state
        out[(size_t)bb * T_SEQ * HDIM + (size_t)t * HDIM + c] = v;      // outputs[B,T,H]
        if (t == T_SEQ - 1)
          out[(size_t)BATCH * T_SEQ * HDIM + (size_t)bb * HDIM + c] = v; // h_final
      }
    }
    gbar(cnt, gen);
  }
}

extern "C" void kernel_launch(void* const* d_in, const int* in_sizes, int n_in,
                              void* d_out, int out_size, void* d_ws, size_t ws_size,
                              hipStream_t stream) {
  const float* x  = (const float*)d_in[0];
  const float* Af = (const float*)d_in[1];
  const float* Bf = (const float*)d_in[2];
  const float* Cf = (const float*)d_in[3];
  const float* Df = (const float*)d_in[4];
  float* out = (float*)d_out;
  (void)in_sizes; (void)n_in; (void)out_size; (void)ws_size;

  char* ws = (char*)d_ws;
  size_t off = 0;
  auto take = [&](size_t bytes) {
    char* p = ws + off;
    off += (bytes + 255) & ~(size_t)255;
    return p;
  };
  unsigned* cnt = (unsigned*)take(256);
  unsigned* gen = cnt + 1;
  bf16_t* Aw = (bf16_t*)take((size_t)NDIM * NDIM * 2);
  bf16_t* Bw = (bf16_t*)take((size_t)NDIM * PDIM * 2);
  bf16_t* Cw = (bf16_t*)take((size_t)HDIM * NDIM * 2);
  bf16_t* Dw = (bf16_t*)take((size_t)HDIM * PDIM * 2);
  bf16_t* u  = (bf16_t*)take((size_t)BATCH * PDIM * 2);
  bf16_t* z  = (bf16_t*)take((size_t)2 * BATCH * NDIM * 2);
  bf16_t* h  = (bf16_t*)take((size_t)BATCH * HDIM * 2);

  hipLaunchKernelGGL(init_barrier_kernel, dim3(1), dim3(32), 0, stream, cnt, gen);

  auto cvt = [&](const float* s, bf16_t* d, int n) {
    hipLaunchKernelGGL(cvt_f32_bf16_kernel, dim3((n + 255) / 256), dim3(256), 0, stream,
                       s, d, n);
  };
  cvt(Af, Aw, NDIM * NDIM);
  cvt(Bf, Bw, NDIM * PDIM);
  cvt(Cf, Cw, HDIM * NDIM);
  cvt(Df, Dw, HDIM * PDIM);

  hipLaunchKernelGGL(implicit_rnn_kernel, dim3(NWG), dim3(TPB), 0, stream,
                     x, out, Aw, Bw, Cw, Dw, u, z, h, cnt, gen);
}
